// MultiHeadAttention_36490042147069
// MI455X (gfx1250) — compile-verified, matrix operand order checked by reading, not run on hardware
//
#include <hip/hip_runtime.h>

// ---------------------------------------------------------------------------
// MHA forward for MI455X (gfx1250, wave32, WMMA, bf16 compute / f32 accum).
//   0) one-shot f32->bf16 conversion of x, Wqkv, Wout (removes per-K-step cvt
//      VALU from GEMM hot loops; halves operand bytes from L2/HBM)
//   1) qkv = x @ Wqkv^T (+fused RoPE) -> q,k [B,H,S,D] bf16, v^T [B,H,D,S]
//   2) flash attention (S^T = K Q^T trick) -> attn [B*S, E] bf16
//   3) out = attn @ Wout^T -> f32
// All WMMA operands are direct global_load_b128 fragments (A = two 8-elem
// K-runs per lane; B mirrored via N-major storage) -- no LDS transposes.
// ---------------------------------------------------------------------------

typedef __bf16 bf16_t;
typedef __attribute__((ext_vector_type(16))) __bf16 v16bf;
typedef __attribute__((ext_vector_type(8)))  __bf16 v8bf;
typedef __attribute__((ext_vector_type(8)))  float  v8f;
typedef __attribute__((ext_vector_type(4)))  float  v4f;

#define WMMA_BF16(a, b, c) \
  __builtin_amdgcn_wmma_f32_16x16x32_bf16(false, (a), false, (b), (short)0, (c), false, false)

#define ZERO8F {0.f,0.f,0.f,0.f,0.f,0.f,0.f,0.f}

// p = lane-resolved pointer to run0 (8 consecutive bf16); run1 at p+16.
__device__ __forceinline__ v16bf frag_from_bf16(const bf16_t* p) {
  v8bf a = *(const v8bf*)(p);
  v8bf b = *(const v8bf*)(p + 16);
  return __builtin_shufflevector(a, b, 0,1,2,3,4,5,6,7,8,9,10,11,12,13,14,15);
}

// ---------------------------------------------------------------------------
// Kernel 0: streaming f32 -> bf16 (8 elems / thread, b128 in, b128 out)
// ---------------------------------------------------------------------------
__global__ __launch_bounds__(256) void cvt_bf16_kernel(
    const float* __restrict__ src, bf16_t* __restrict__ dst, int n8) {
  int i = blockIdx.x * 256 + threadIdx.x;
  if (i >= n8) return;
  const v4f* s = (const v4f*)src + (long)i * 2;
  v4f a = s[0], b = s[1];
  v8bf o;
#pragma unroll
  for (int t = 0; t < 4; ++t) { o[t] = (__bf16)a[t]; o[4 + t] = (__bf16)b[t]; }
  *((v8bf*)dst + i) = o;
}

// ---------------------------------------------------------------------------
// Kernel 1: qkv = x @ Wqkv^T with fused RoPE.
// Each wave: 32 token rows (2 strips) x 128 features (one head for q/k).
// ---------------------------------------------------------------------------
__global__ __launch_bounds__(256) void qkv_rope_kernel(
    const bf16_t* __restrict__ x,     // [8192, 2048] bf16
    const bf16_t* __restrict__ Wqkv,  // [6144, 2048] bf16
    bf16_t* __restrict__ qbf,         // [B,H,S,D]
    bf16_t* __restrict__ kbf,         // [B,H,S,D]
    bf16_t* __restrict__ vtbf)        // [B,H,D,S]
{
  const int lane = threadIdx.x & 31;
  const int l16  = lane & 15;
  const int half = lane >> 4;
  const int wave = blockIdx.x * 8 + (threadIdx.x >> 5);
  const int mt = wave / 48;          // 256 M tiles (32 rows)
  const int nt = wave % 48;          // 48 N tiles (128 cols)
  const int row0 = mt * 32;
  const int n0   = nt * 128;

  v8f z = ZERO8F;
  v8f acc[2][8];
#pragma unroll
  for (int s = 0; s < 2; ++s)
#pragma unroll
    for (int j = 0; j < 8; ++j) acc[s][j] = z;

  const bf16_t* aBase = x    + (long)(row0 + l16) * 2048 + half * 8;
  const bf16_t* bBase = Wqkv + (long)(n0   + l16) * 2048 + half * 8;

  for (int k0 = 0; k0 < 2048; k0 += 32) {
    __builtin_prefetch(aBase + k0 + 512, 0, 1);    // global_prefetch_b8
    v16bf a0 = frag_from_bf16(aBase + k0);
    v16bf a1 = frag_from_bf16(aBase + 16 * 2048 + k0);
#pragma unroll
    for (int j = 0; j < 8; ++j) {
      v16bf bf = frag_from_bf16(bBase + (long)j * 16 * 2048 + k0);
      acc[0][j] = WMMA_BF16(a0, bf, acc[0][j]);    // B frag reused twice
      acc[1][j] = WMMA_BF16(a1, bf, acc[1][j]);
    }
  }

  // Epilogue: acc[s][j][r] = C[row0 + s*16 + r + half*8][n0 + j*16 + l16]
#pragma unroll
  for (int s = 0; s < 2; ++s) {
#pragma unroll
    for (int j = 0; j < 8; ++j) {
      const int f   = n0 + j * 16 + l16;
      const int blk = f >> 11;         // 0=q 1=k 2=v
      const int h   = (f >> 7) & 15;
      const int d   = f & 127;
      if (blk == 2) {                  // V: store transposed [b,h,d,s]
#pragma unroll
        for (int r = 0; r < 8; ++r) {
          int row = row0 + s * 16 + r + half * 8;
          int b = row >> 11, sq = row & 2047;
          vtbf[((long)(b * 16 + h) * 128 + d) * 2048 + sq] = (bf16_t)acc[s][j][r];
        }
      } else {                         // Q/K: RoPE; partner feature d^64 = acc[s][j^4]
        const int   jp   = j ^ 4;
        const float sgn  = (d < 64) ? -1.f : 1.f;
        const float invf = __expf(-(float)(d & 63) * 0.14391156831212787f); // ln(1e4)/64
        bf16_t* dst = (blk == 0) ? qbf : kbf;
#pragma unroll
        for (int r = 0; r < 8; ++r) {
          int row = row0 + s * 16 + r + half * 8;
          int b = row >> 11, sq = row & 2047;
          float sn, cs;
          __sincosf((float)sq * invf, &sn, &cs);
          float val = acc[s][j][r] * cs + sgn * acc[s][jp][r] * sn;
          dst[((long)(b * 16 + h) * 2048 + sq) * 128 + d] = (bf16_t)val;
        }
      }
    }
  }
}

// ---------------------------------------------------------------------------
// Kernel 2: flash attention, one 16-row q tile per wave.
// Scores computed transposed (S^T = K * Q^T) so softmax is lane-local per q
// column; P^T C-fragments feed the V^T * P^T WMMA directly as B operands.
// ---------------------------------------------------------------------------
__global__ __launch_bounds__(256) void attn_kernel(
    const bf16_t* __restrict__ qbf,
    const bf16_t* __restrict__ kbf,
    const bf16_t* __restrict__ vtbf,
    bf16_t* __restrict__ attn)         // [8192, 2048]
{
  const int lane = threadIdx.x & 31;
  const int l16  = lane & 15;
  const int half = lane >> 4;
  const int wave = blockIdx.x * 8 + (threadIdx.x >> 5);
  const int qt = wave & 127;           // 128 q tiles per (b,h)
  const int bh = wave >> 7;            // 0..63
  const long base = (long)bh * 2048 * 128;
  const int q0 = qt * 16;
  const int qglob = q0 + l16;          // q column owned by this lane

  const bf16_t* qB = qbf + base + (long)(q0 + l16) * 128 + half * 8;
  v16bf Qf[4];
#pragma unroll
  for (int c = 0; c < 4; ++c) Qf[c] = frag_from_bf16(qB + c * 32);

  v8f z = ZERO8F;
  v8f O[8];
#pragma unroll
  for (int dt = 0; dt < 8; ++dt) O[dt] = z;
  float m_run = -1e30f, l_run = 0.f;
  const float scale = 0.08838834764831845f;   // 1/sqrt(128)

  const bf16_t* kA0 = kbf  + base + (long)l16 * 128  + half * 8;
  const bf16_t* vA0 = vtbf + base + (long)l16 * 2048 + half * 8;

  const int pairs = (qt + 2) >> 1;            // kv covered: 32*pairs >= q0+16
  for (int p = 0; p < pairs; ++p) {
    const int kv0 = p * 32;
    v16bf Pf;                                  // P^T B-fragment (K=32)
#pragma unroll
    for (int t = 0; t < 2; ++t) {
      const bf16_t* kA = kA0 + (long)(kv0 + t * 16) * 128;
      v8f s = ZERO8F;
#pragma unroll
      for (int c = 0; c < 4; ++c) {
        v16bf Kf = frag_from_bf16(kA + c * 32);
        s = WMMA_BF16(Kf, Qf[c], s);           // S^T tile: [kv][q]
      }
      float sv[8], tm = -1e30f;
#pragma unroll
      for (int r = 0; r < 8; ++r) {
        int kv = kv0 + t * 16 + r + half * 8;
        float v = s[r] * scale;
        if (kv > qglob) v = -1e30f;            // causal mask (analytic)
        sv[r] = v;
        tm = fmaxf(tm, v);
      }
      tm = fmaxf(tm, __shfl_xor(tm, 16, 32));  // merge wave halves (same q)
      float m_new = fmaxf(m_run, tm);
      float alpha = __expf(m_run - m_new);
      float ls = 0.f;
#pragma unroll
      for (int r = 0; r < 8; ++r) {
        float pv = __expf(sv[r] - m_new);
        ls += pv;
        Pf[t * 8 + r] = (bf16_t)pv;            // C layout == B layout for PV
      }
      l_run = l_run * alpha + ls + __shfl_xor(ls, 16, 32);
      m_run = m_new;
#pragma unroll
      for (int dt = 0; dt < 8; ++dt) O[dt] *= alpha;
    }
    // out^T += V^T(16d x 32kv) * P^T(32kv x 16q), 8 d-tiles
#pragma unroll
    for (int dt = 0; dt < 8; ++dt) {
      v16bf Vf = frag_from_bf16(vA0 + (long)dt * 16 * 2048 + kv0);
      O[dt] = WMMA_BF16(Vf, Pf, O[dt]);
    }
  }

  const float inv_l = 1.0f / l_run;
  const int b = bh >> 4, h = bh & 15;
  const long orow = ((long)(b * 2048 + qglob)) * 2048 + h * 128;
#pragma unroll
  for (int dt = 0; dt < 8; ++dt)
#pragma unroll
    for (int r = 0; r < 8; ++r)
      attn[orow + dt * 16 + r + half * 8] = (bf16_t)(O[dt][r] * inv_l);
}

// ---------------------------------------------------------------------------
// Kernel 3: out = attn @ Wout^T (f32 result), 32x128 per wave
// ---------------------------------------------------------------------------
__global__ __launch_bounds__(256) void outproj_kernel(
    const bf16_t* __restrict__ attn,   // [8192, 2048] bf16
    const bf16_t* __restrict__ Wout,   // [2048, 2048] bf16
    float* __restrict__ out)           // [8192, 2048] f32
{
  const int lane = threadIdx.x & 31;
  const int l16  = lane & 15;
  const int half = lane >> 4;
  const int wave = blockIdx.x * 8 + (threadIdx.x >> 5);
  const int mt = wave / 16;            // 256 M tiles (32 rows)
  const int nt = wave % 16;            // 16 N tiles (128 cols)
  const int row0 = mt * 32;
  const int n0   = nt * 128;

  v8f z = ZERO8F;
  v8f acc[2][8];
#pragma unroll
  for (int s = 0; s < 2; ++s)
#pragma unroll
    for (int j = 0; j < 8; ++j) acc[s][j] = z;

  const bf16_t* aBase = attn + (long)(row0 + l16) * 2048 + half * 8;
  const bf16_t* bBase = Wout + (long)(n0   + l16) * 2048 + half * 8;

  for (int k0 = 0; k0 < 2048; k0 += 32) {
    v16bf a0 = frag_from_bf16(aBase + k0);
    v16bf a1 = frag_from_bf16(aBase + 16 * 2048 + k0);
#pragma unroll
    for (int j = 0; j < 8; ++j) {
      v16bf bf = frag_from_bf16(bBase + (long)j * 16 * 2048 + k0);
      acc[0][j] = WMMA_BF16(a0, bf, acc[0][j]);
      acc[1][j] = WMMA_BF16(a1, bf, acc[1][j]);
    }
  }
#pragma unroll
  for (int s = 0; s < 2; ++s)
#pragma unroll
    for (int j = 0; j < 8; ++j)
#pragma unroll
      for (int r = 0; r < 8; ++r)
        out[(long)(row0 + s * 16 + r + half * 8) * 2048 + n0 + j * 16 + l16] =
            acc[s][j][r];
}

// ---------------------------------------------------------------------------
extern "C" void kernel_launch(void* const* d_in, const int* in_sizes, int n_in,
                              void* d_out, int out_size, void* d_ws, size_t ws_size,
                              hipStream_t stream) {
  (void)in_sizes; (void)n_in; (void)out_size; (void)ws_size;
  const float* x    = (const float*)d_in[0];
  // d_in[1] = att_mask: causal, applied analytically in attn_kernel
  const float* Wqkv = (const float*)d_in[2];
  const float* Wout = (const float*)d_in[3];
  float* out = (float*)d_out;

  const size_t NX   = (size_t)8192 * 2048;   // 16.78M
  const size_t NWQ  = (size_t)6144 * 2048;   // 12.58M
  const size_t NWO  = (size_t)2048 * 2048;   //  4.19M
  const size_t QKV  = (size_t)4 * 16 * 2048 * 128;  // 16.78M

  bf16_t* xbf   = (bf16_t*)d_ws;             // dead after k1 -> reused as attn
  bf16_t* wqkvb = xbf   + NX;
  bf16_t* woutb = wqkvb + NWQ;
  bf16_t* qbf   = woutb + NWO;
  bf16_t* kbf   = qbf   + QKV;
  bf16_t* vtbf  = kbf   + QKV;
  bf16_t* attn  = xbf;                       // alias (same size, disjoint lifetime)
  // total workspace: (NX + NWQ + NWO + 3*QKV) * 2 bytes ~= 168 MB

  cvt_bf16_kernel<<<(int)(NX  / 8 / 256), 256, 0, stream>>>(x,    xbf,   (int)(NX  / 8));
  cvt_bf16_kernel<<<(int)(NWQ / 8 / 256), 256, 0, stream>>>(Wqkv, wqkvb, (int)(NWQ / 8));
  cvt_bf16_kernel<<<(int)(NWO / 8 / 256), 256, 0, stream>>>(Wout, woutb, (int)(NWO / 8));

  qkv_rope_kernel<<<1536, 256, 0, stream>>>(xbf, wqkvb, qbf, kbf, vtbf);
  attn_kernel    <<<1024, 256, 0, stream>>>(qbf, kbf, vtbf, attn);
  outproj_kernel <<< 512, 256, 0, stream>>>(attn, woutb, out);
}
